// LlamaAttention_50113678410517
// MI455X (gfx1250) — matrix operand order repaired and checked by reading, not compiled, not run on hardware
//
#include <hip/hip_runtime.h>
#include <hip/hip_bf16.h>

typedef __bf16 bf16_t;
typedef __attribute__((ext_vector_type(16))) __bf16 v16bf;
typedef __attribute__((ext_vector_type(8)))  float  v8f;
typedef int si32x4 __attribute__((ext_vector_type(4)));
typedef int si32x8 __attribute__((ext_vector_type(8)));

union Frag16 { v16bf v; uint4 q[2]; bf16_t h[16]; };

#define B_SZ 2
#define L_SZ 2048
#define DM   2048
#define NH   32
#define NKV  8
#define DH   64
#define KVD  512

// ---------------------------------------------------------------- DPP 16-lane butterfly
// Row stats live across 16-lane halves (lane>>4 fixed): xor-1/2 via quad_perm,
// xor-4/8 via row_xmask -> pure VALU, no ds_bpermute / dscnt stalls.
template <int CTRL>
__device__ __forceinline__ float dpp_xor_f(float x) {
    union { float f; int i; } u, r;
    u.f = x;
    r.i = __builtin_amdgcn_mov_dpp(u.i, CTRL, 0xF, 0xF, true);
    return r.f;
}
__device__ __forceinline__ float rowmax16(float x) {
    x = fmaxf(x, dpp_xor_f<0xB1>(x));    // quad_perm [1,0,3,2]  (xor 1)
    x = fmaxf(x, dpp_xor_f<0x4E>(x));    // quad_perm [2,3,0,1]  (xor 2)
    x = fmaxf(x, dpp_xor_f<0x164>(x));   // row_xmask:4
    x = fmaxf(x, dpp_xor_f<0x168>(x));   // row_xmask:8
    return x;
}
__device__ __forceinline__ float rowsum16(float x) {
    x += dpp_xor_f<0xB1>(x);
    x += dpp_xor_f<0x4E>(x);
    x += dpp_xor_f<0x164>(x);
    x += dpp_xor_f<0x168>(x);
    return x;
}

// ---------------------------------------------------------------- fp32 -> bf16 (row-major copy)
__global__ void cvt_f32_bf16(const float* __restrict__ s, bf16_t* __restrict__ d, long n) {
    long i = (long)blockIdx.x * blockDim.x + threadIdx.x;
    long stride = (long)gridDim.x * blockDim.x;
    for (; i < n; i += stride) d[i] = (bf16_t)s[i];
}

// ---------------------------------------------------------------- fp32 -> bf16 transposed
__global__ __launch_bounds__(256) void cvt_transpose_bf16(
    const float* __restrict__ src, bf16_t* __restrict__ dst, int R, int C)
{
    __shared__ float tile[32][33];
    int c0 = blockIdx.x * 32, r0 = blockIdx.y * 32;
    int tx = threadIdx.x, ty = threadIdx.y;          // (32, 8)
    #pragma unroll
    for (int i = 0; i < 32; i += 8)
        tile[ty + i][tx] = src[(long)(r0 + ty + i) * C + c0 + tx];
    __syncthreads();
    #pragma unroll
    for (int i = 0; i < 32; i += 8)
        dst[(long)(c0 + ty + i) * R + r0 + tx] = (bf16_t)tile[tx][ty + i];
}

// ---------------------------------------------------------------- WMMA GEMM + TDM
// C(MxN) = A(MxK bf16 row) @ B, Bt pre-transposed (N x K row-major).
// A tile (128x32) DMA'd to LDS by the Tensor Data Mover, double-buffered on
// TENSORcnt (issue t+1, wait tensorcnt<=1 for t).  Pad 4DW/16DW -> 80B pitch.
template <typename OutT, bool TRANSP_OUT>
__global__ __launch_bounds__(256) void gemm_bf16_wmma(
    const bf16_t* __restrict__ A, const bf16_t* __restrict__ Bt,
    OutT* __restrict__ C, int M, int N, int K)
{
    constexpr int BM = 128, BN = 128, BK = 32;
    __shared__ __align__(16) bf16_t As[2][BM][BK + 8];

    const int tid  = threadIdx.x;
    const int lane = tid & 31;
    const int w    = tid >> 5;
    const int wm   = w >> 1;              // 0..3
    const int wn   = w & 1;               // 0..1
    const int g    = lane >> 4, m = lane & 15;
    const int bm   = blockIdx.y * BM;
    const int bn   = blockIdx.x * BN;

    si32x8 g1;
    g1[0] = (1 << 16) | (1 << 20) | (3 << 22) | (3 << 25);  // 2B elems, pad_en, 16DW intv, 4DW pad
    g1[1] = (K & 0xFFFF) << 16;
    g1[2] = (int)(((unsigned)K >> 16) | ((unsigned)(M & 0xFFFF) << 16));
    g1[3] = (int)(((unsigned)M >> 16) | ((unsigned)BK << 16));
    g1[4] = BM;
    g1[5] = K;
    g1[6] = 0;
    g1[7] = 0;

    auto tdm_issue = [&](int buf, int k0) {
        unsigned long long ga = (unsigned long long)(const void*)&A[(long)bm * K + k0];
        unsigned lds = (unsigned)(unsigned long long)&As[buf][0][0];
        si32x4 g0;
        g0[0] = 1;
        g0[1] = (int)lds;
        g0[2] = (int)(unsigned)(ga & 0xFFFFFFFFu);
        g0[3] = (int)(((unsigned)(ga >> 32) & 0x1FFFFFFu) | (2u << 30));
        asm volatile("tensor_load_to_lds %0, %1" :: "s"(g0), "s"(g1) : "memory");
    };

    v8f acc[2][4] = {};

    if (w == 0) tdm_issue(0, 0);
    const int T = K / BK;
    for (int t = 0; t < T; ++t) {
        const int k0 = t * BK;
        if (w == 0) {
            if (t + 1 < T) {
                tdm_issue((t + 1) & 1, k0 + BK);
                __builtin_amdgcn_s_wait_tensorcnt(1);
            } else {
                __builtin_amdgcn_s_wait_tensorcnt(0);
            }
        }
        __syncthreads();
        const bf16_t (*At)[BK + 8] = As[t & 1];

        Frag16 a[2], b[4];
        #pragma unroll
        for (int mi = 0; mi < 2; ++mi) {
            const bf16_t* row = &At[wm * 32 + mi * 16 + m][0];
            a[mi].q[0] = *(const uint4*)(row + 8 * g);
            a[mi].q[1] = *(const uint4*)(row + 16 + 8 * g);
        }
        #pragma unroll
        for (int ni = 0; ni < 4; ++ni) {
            const bf16_t* brow = Bt + (long)(bn + wn * 64 + ni * 16 + m) * K + k0;
            b[ni].q[0] = *(const uint4*)(brow + 8 * g);
            b[ni].q[1] = *(const uint4*)(brow + 16 + 8 * g);
        }
        #pragma unroll
        for (int mi = 0; mi < 2; ++mi)
            #pragma unroll
            for (int ni = 0; ni < 4; ++ni)
                acc[mi][ni] = __builtin_amdgcn_wmma_f32_16x16x32_bf16(
                    false, a[mi].v, false, b[ni].v, (short)0, acc[mi][ni], false, false);
        __syncthreads();
    }

    #pragma unroll
    for (int mi = 0; mi < 2; ++mi)
        #pragma unroll
        for (int ni = 0; ni < 4; ++ni)
            #pragma unroll
            for (int r = 0; r < 8; ++r) {
                int row = bm + wm * 32 + mi * 16 + 8 * g + r;
                int col = bn + wn * 64 + ni * 16 + m;
                if (TRANSP_OUT) C[(long)col * M + row] = (OutT)acc[mi][ni][r];
                else            C[(long)row * N + col] = (OutT)acc[mi][ni][r];
            }
}

// ---------------------------------------------------------------- RoPE (in-place, optional scale)
__global__ void rope_apply(bf16_t* __restrict__ T, const float* __restrict__ cosT,
                           const float* __restrict__ sinT, int nHeads, float scale)
{
    int idx = blockIdx.x * blockDim.x + threadIdx.x;
    int total = B_SZ * L_SZ * nHeads * (DH / 2);
    if (idx >= total) return;
    int d   = idx & 31;
    int h   = (idx >> 5) % nHeads;
    int tok = idx / (32 * nHeads);
    int l   = tok % L_SZ;
    long base = (long)tok * nHeads * DH + h * DH;
    float x1 = (float)T[base + d];
    float x2 = (float)T[base + d + 32];
    float c1 = cosT[l * DH + d],      s1 = sinT[l * DH + d];
    float c2 = cosT[l * DH + d + 32], s2 = sinT[l * DH + d + 32];
    T[base + d]      = (bf16_t)((x1 * c1 - x2 * s1) * scale);
    T[base + d + 32] = (bf16_t)((x2 * c2 + x1 * s2) * scale);
}

// ---------------------------------------------------------------- Flash attention
// grid = (L/64, NH, B); 128 threads = 4 waves, 16 q-rows each.  128-key blocks:
// K and V^T tiles async-staged to LDS once per block (shared by all 4 waves),
// one softmax update per 128 keys, DPP-only row reductions.  Fully-masked
// 16-key sub-tiles skip their WMMAs.  1/8 scale pre-folded into Q.
__global__ __launch_bounds__(128) void flash_attn_wmma(
    const bf16_t* __restrict__ Q, const bf16_t* __restrict__ Kc,
    const bf16_t* __restrict__ Vt, bf16_t* __restrict__ O)
{
    constexpr int KB = 128;
    __shared__ __align__(16) bf16_t Ks[KB][72];         // [key][dh]
    __shared__ __align__(16) bf16_t Vs[64][KB + 8];     // [dh][key]
    __shared__ __align__(16) bf16_t Pst[4][16][KB + 8]; // per-wave P reshape
    const int tid  = threadIdx.x;
    const int lane = tid & 31;
    const int wave = tid >> 5;
    const int g = lane >> 4, m = lane & 15;
    const int qt  = blockIdx.x;
    const int h   = blockIdx.y;
    const int b   = blockIdx.z;
    const int kvh = h >> 2;                             // GQA 4:1
    const int Mrows = B_SZ * L_SZ;

    const bf16_t* Qp = Q  + ((long)b * L_SZ) * DM  + h   * DH;
    const bf16_t* Kp = Kc + ((long)b * L_SZ) * KVD + kvh * DH;
    const bf16_t* Vp = Vt + (long)kvh * 64 * Mrows + (long)b * L_SZ;  // [dh][token]

    const int qbase = qt * 64 + wave * 16;

    Frag16 qf[2];
    {
        const bf16_t* qrow = Qp + (long)(qbase + m) * DM;
        #pragma unroll
        for (int s = 0; s < 2; ++s) {
            qf[s].q[0] = *(const uint4*)(qrow + s * 32 + 8 * g);
            qf[s].q[1] = *(const uint4*)(qrow + s * 32 + 16 + 8 * g);
        }
    }

    v8f o[4] = {};
    float mrow[8], lrow[8];
    #pragma unroll
    for (int r = 0; r < 8; ++r) { mrow[r] = -1e30f; lrow[r] = 0.f; }

    const int nblocks = (qt >> 1) + 1;                  // causal, 128-key blocks
    for (int j = 0; j < nblocks; ++j) {
        const int kb = j * KB;
        __syncthreads();                                // prior readers done
        // async-stage K [128 x 64] and V^T [64 x 128]
        #pragma unroll
        for (int i = 0; i < 8; ++i) {
            int idx = tid + i * 128;                    // 1024 b128 each
            {
                int r = idx >> 3, c = (idx & 7) * 8;
                unsigned lds_addr = (unsigned)(unsigned long long)&Ks[r][c];
                const bf16_t* gp = Kp + (long)(kb + r) * KVD + c;
                asm volatile("global_load_async_to_lds_b128 %0, %1, off"
                             :: "v"(lds_addr), "v"(gp) : "memory");
            }
            {
                int r = idx >> 4, c = (idx & 15) * 8;
                unsigned lds_addr = (unsigned)(unsigned long long)&Vs[r][c];
                const bf16_t* gp = Vp + (long)r * Mrows + kb + c;
                asm volatile("global_load_async_to_lds_b128 %0, %1, off"
                             :: "v"(lds_addr), "v"(gp) : "memory");
            }
        }
        asm volatile("s_wait_asynccnt 0x0" ::: "memory");
        __syncthreads();

        // S = Q @ K^T (8 key sub-tiles of 16)
        v8f st[8];
        #pragma unroll
        for (int nt = 0; nt < 8; ++nt) {
            if (kb + nt * 16 <= qbase + 15) {           // tile has live keys
                v8f s = {};
                #pragma unroll
                for (int ss = 0; ss < 2; ++ss) {
                    Frag16 kf;
                    const bf16_t* krow = &Ks[nt * 16 + m][0];
                    kf.q[0] = *(const uint4*)(krow + ss * 32 + 8 * g);
                    kf.q[1] = *(const uint4*)(krow + ss * 32 + 16 + 8 * g);
                    s = __builtin_amdgcn_wmma_f32_16x16x32_bf16(
                            false, qf[ss].v, false, kf.v, (short)0, s, false, false);
                }
                #pragma unroll
                for (int r = 0; r < 8; ++r) {           // causal mask
                    int row = qbase + 8 * g + r;
                    int key = kb + nt * 16 + m;
                    s[r] = (key > row) ? -1e30f : s[r];
                }
                st[nt] = s;
            } else {
                #pragma unroll
                for (int r = 0; r < 8; ++r) st[nt][r] = -1e30f;
            }
        }

        // online softmax, DPP-only reductions
        #pragma unroll
        for (int r = 0; r < 8; ++r) {
            float mx = st[0][r];
            #pragma unroll
            for (int nt = 1; nt < 8; ++nt) mx = fmaxf(mx, st[nt][r]);
            mx = rowmax16(mx);
            float mnew  = fmaxf(mrow[r], mx);
            float scale = __expf(mrow[r] - mnew);
            float psum  = 0.f;
            #pragma unroll
            for (int nt = 0; nt < 8; ++nt) {
                float p = __expf(st[nt][r] - mnew);
                st[nt][r] = p;
                psum += p;
            }
            psum = rowsum16(psum);
            lrow[r] = lrow[r] * scale + psum;
            mrow[r] = mnew;
            #pragma unroll
            for (int dt = 0; dt < 4; ++dt) o[dt][r] *= scale;
        }

        // reshape P (C layout) -> A-fragment layout via per-wave LDS
        #pragma unroll
        for (int nt = 0; nt < 8; ++nt)
            #pragma unroll
            for (int r = 0; r < 8; ++r)
                Pst[wave][8 * g + r][nt * 16 + m] = (bf16_t)st[nt][r];
        asm volatile("" ::: "memory");                  // DS in-order within wave

        // O += P @ V  (contraction over 128 keys = 4 wmma k-steps)
        #pragma unroll
        for (int dt = 0; dt < 4; ++dt) {
            #pragma unroll
            for (int ss = 0; ss < 4; ++ss) {
                Frag16 pf, vf;
                const bf16_t* prow = &Pst[wave][m][0];
                pf.q[0] = *(const uint4*)(prow + ss * 32 + 8 * g);
                pf.q[1] = *(const uint4*)(prow + ss * 32 + 16 + 8 * g);
                const bf16_t* vrow = &Vs[dt * 16 + m][0];
                vf.q[0] = *(const uint4*)(vrow + ss * 32 + 8 * g);
                vf.q[1] = *(const uint4*)(vrow + ss * 32 + 16 + 8 * g);
                o[dt] = __builtin_amdgcn_wmma_f32_16x16x32_bf16(
                            false, pf.v, false, vf.v, (short)0, o[dt], false, false);
            }
        }
        asm volatile("" ::: "memory");
    }

    // normalize and store (B, L, H, Dh) == (B*L, 2048) bf16
    #pragma unroll
    for (int r = 0; r < 8; ++r) {
        float inv = 1.0f / lrow[r];
        int row = qbase + 8 * g + r;
        bf16_t* orow = O + ((long)b * L_SZ + row) * DM + h * DH;
        #pragma unroll
        for (int dt = 0; dt < 4; ++dt)
            orow[dt * 16 + m] = (bf16_t)(o[dt][r] * inv);
    }
}

// ---------------------------------------------------------------- launch
extern "C" void kernel_launch(void* const* d_in, const int* in_sizes, int n_in,
                              void* d_out, int out_size, void* d_ws, size_t ws_size,
                              hipStream_t stream)
{
    (void)in_sizes; (void)n_in; (void)out_size; (void)ws_size;
    const float* x    = (const float*)d_in[0];
    const float* cosT = (const float*)d_in[1];
    const float* sinT = (const float*)d_in[2];
    /* d_in[3] = mask: unused, causal mask applied analytically */
    const float* Wq   = (const float*)d_in[4];
    const float* Wk   = (const float*)d_in[5];
    const float* Wv   = (const float*)d_in[6];
    const float* Wo   = (const float*)d_in[7];
    float* out = (float*)d_out;

    char* ws = (char*)d_ws;
    const long MB = 1024 * 1024;
    bf16_t* xb  = (bf16_t*)(ws +  0 * MB);  // 16 MB  x bf16 (row-major)
    bf16_t* Wqt = (bf16_t*)(ws + 16 * MB);  //  8 MB  Wq^T (N x K)
    bf16_t* Wkt = (bf16_t*)(ws + 24 * MB);  //  2 MB  Wk^T
    bf16_t* Wvt = (bf16_t*)(ws + 26 * MB);  //  2 MB  Wv^T
    bf16_t* Wot = (bf16_t*)(ws + 28 * MB);  //  8 MB  Wo^T
    bf16_t* Qb  = (bf16_t*)(ws + 36 * MB);  // 16 MB  Q (tokens x 2048)
    bf16_t* Kb  = (bf16_t*)(ws + 52 * MB);  //  4 MB  K (tokens x 512)
    bf16_t* Vtb = (bf16_t*)(ws + 56 * MB);  //  4 MB  V^T (512 x tokens)
    bf16_t* Ab  = (bf16_t*)(ws + 60 * MB);  // 16 MB  attn out (tokens x 2048)

    const int Mrows = B_SZ * L_SZ;          // 4096

    cvt_f32_bf16<<<2048, 256, 0, stream>>>(x, xb, (long)Mrows * DM);
    cvt_transpose_bf16<<<dim3(DM  / 32, DM / 32), dim3(32, 8), 0, stream>>>(Wq, Wqt, DM, DM);
    cvt_transpose_bf16<<<dim3(KVD / 32, DM / 32), dim3(32, 8), 0, stream>>>(Wk, Wkt, DM, KVD);
    cvt_transpose_bf16<<<dim3(KVD / 32, DM / 32), dim3(32, 8), 0, stream>>>(Wv, Wvt, DM, KVD);
    cvt_transpose_bf16<<<dim3(DM  / 32, DM / 32), dim3(32, 8), 0, stream>>>(Wo, Wot, DM, DM);

    gemm_bf16_wmma<bf16_t, false><<<dim3(DM  / 128, Mrows / 128), 256, 0, stream>>>(xb, Wqt, Qb,  Mrows, DM,  DM);
    gemm_bf16_wmma<bf16_t, false><<<dim3(KVD / 128, Mrows / 128), 256, 0, stream>>>(xb, Wkt, Kb,  Mrows, KVD, DM);
    gemm_bf16_wmma<bf16_t, true ><<<dim3(KVD / 128, Mrows / 128), 256, 0, stream>>>(xb, Wvt, Vtb, Mrows, KVD, DM);

    rope_apply<<<(Mrows * NH  * 32 + 255) / 256, 256, 0, stream>>>(Qb, cosT, sinT, NH, 0.125f);
    rope_apply<<<(Mrows * NKV * 32 + 255) / 256, 256, 0, stream>>>(Kb, cosT, sinT, NKV, 1.0f);

    flash_attn_wmma<<<dim3(L_SZ / 64, NH, B_SZ), 128, 0, stream>>>(Qb, Kb, Vtb, Ab);

    gemm_bf16_wmma<float, false><<<dim3(DM / 128, Mrows / 128), 256, 0, stream>>>(Ab, Wot, out, Mrows, DM, DM);
}
